// Compressor_10230612099841
// MI455X (gfx1250) — matrix operand order repaired and checked by reading, not compiled, not run on hardware
//
#include <hip/hip_runtime.h>
#include <math.h>
#include <stdint.h>

// ---------------------------------------------------------------------------
// MI455X (gfx1250) implementation: bf16 WMMA (fp32 accum) for all matmuls,
// flash-attention with online softmax, fused SwiGLU GEMM, async global->LDS
// staging (ASYNCcnt) where no format conversion is needed, fast-rcp epilogues.
// ---------------------------------------------------------------------------

typedef __attribute__((ext_vector_type(16))) __bf16 v16bf;
typedef __attribute__((ext_vector_type(8)))  __bf16 v8bf;
typedef __attribute__((ext_vector_type(4)))  __bf16 v4bf;
typedef __attribute__((ext_vector_type(8)))  float  v8f;

union V16 { v16bf v; v8bf h[2]; };

static constexpr int Bsz   = 2;
static constexpr int Lseq  = 2048;
static constexpr int Dm    = 1024;
static constexpr int Hh    = 16;
static constexpr int Dhd   = 64;
static constexpr int Ff    = 4096;
static constexpr int Mrows = Bsz * Lseq;      // 4096
static constexpr int MAXC  = 32;

#define DEV __device__ __forceinline__

DEV v8f wmma_bf16(v16bf a, v16bf b, v8f c) {
  // D = A(16x32 bf16) * B(32x16 bf16) + C(16x16 f32)
  return __builtin_amdgcn_wmma_f32_16x16x32_bf16(false, a, false, b, (short)0, c,
                                                 false, false);
}

DEV float fast_rcp(float x) { return __builtin_amdgcn_rcpf(x); }
DEV float fast_silu(float v) { return v * fast_rcp(1.0f + __expf(-v)); }

// Async 16B global->LDS copy (tracked by ASYNCcnt). LDS operand is the raw
// LDS byte offset: shared-aperture lives in addr[63:32], so truncating the
// generic pointer to 32 bits gives the LDS address (ISA 10.2).
DEV void async_copy16(const void* lds_ptr, const void* gptr) {
  uint32_t lds = (uint32_t)(uintptr_t)lds_ptr;
  asm volatile("global_load_async_to_lds_b128 %0, %1, off"
               :: "v"(lds), "v"(gptr) : "memory");
}
DEV void async_wait0() { asm volatile("s_wait_asynccnt 0x0" ::: "memory"); }

DEV float wred_max16(float v) {
  v = fmaxf(v, __shfl_xor(v, 1, 32));
  v = fmaxf(v, __shfl_xor(v, 2, 32));
  v = fmaxf(v, __shfl_xor(v, 4, 32));
  v = fmaxf(v, __shfl_xor(v, 8, 32));
  return v;
}
DEV float wred_sum16(float v) {
  v += __shfl_xor(v, 1, 32);
  v += __shfl_xor(v, 2, 32);
  v += __shfl_xor(v, 4, 32);
  v += __shfl_xor(v, 8, 32);
  return v;
}
DEV float wred_sum32(float v) {
  v += __shfl_xor(v, 1, 32);
  v += __shfl_xor(v, 2, 32);
  v += __shfl_xor(v, 4, 32);
  v += __shfl_xor(v, 8, 32);
  v += __shfl_xor(v, 16, 32);
  return v;
}

// ---------------------------------------------------------------------------
// Generic tiled GEMM: C[M,N] = epi( A[M,K] * B[K,N] (+bias[N]) (+Res[M,N]) )
// A is fp32 or bf16 (AT), B fp32; both staged to LDS as bf16.
// Block tile 128x128, K-tile 32, 256 threads = 8 waves, wave tile 64x32.
// ---------------------------------------------------------------------------
template <typename AT, bool HAS_BIAS, bool ACT_SILU, bool HAS_RES>
__global__ void __launch_bounds__(256)
gemm_kernel(const AT* __restrict__ A, const float* __restrict__ Bm,
            const float* __restrict__ bias, const float* __restrict__ Res,
            float* __restrict__ C, int M, int N, int K) {
  __shared__ __bf16 As[128][48];   // [m][k], 16B-aligned rows (96B stride)
  __shared__ __bf16 Bs[128][48];   // [n][k] (transposed), 96B stride

  const int tid  = threadIdx.x;
  const int lane = tid & 31;
  const int wave = tid >> 5;
  const int wm   = wave >> 2;      // 0..1  -> row group of 64
  const int wn   = wave & 3;       // 0..3  -> col group of 32
  const int m0   = blockIdx.y * 128;
  const int n0   = blockIdx.x * 128;

  v8f acc[4][2] = {};

  for (int k0 = 0; k0 < K; k0 += 32) {
    __syncthreads();
    // ---- stage A tile (128 x 32) ----
    if constexpr (sizeof(AT) == 4) {
#pragma unroll
      for (int i = 0; i < 4; ++i) {
        int idx = tid + 256 * i;          // 1024 float4 chunks
        int row = idx >> 3;
        int kf  = (idx & 7) * 4;
        const float* gp = &A[(size_t)(m0 + row) * K + k0 + kf];
        float4 t = *(const float4*)gp;
        if (k0 + 32 < K) __builtin_prefetch(gp + 32, 0, 0);
        v4bf pk;
        pk[0] = (__bf16)t.x; pk[1] = (__bf16)t.y;
        pk[2] = (__bf16)t.z; pk[3] = (__bf16)t.w;
        *(v4bf*)&As[row][kf] = pk;
      }
    } else {
      // pure bf16 copy -> async global->LDS (no VGPR round trip)
#pragma unroll
      for (int i = 0; i < 2; ++i) {
        int idx = tid + 256 * i;          // 512 v8bf chunks
        int row = idx >> 2;
        int kf  = (idx & 3) * 8;
        async_copy16(&As[row][kf], &A[(size_t)(m0 + row) * K + k0 + kf]);
      }
    }
    // ---- stage B tile (32 x 128) transposed ----
#pragma unroll
    for (int i = 0; i < 4; ++i) {
      int idx = tid + 256 * i;            // 1024 float4 chunks
      int kr  = idx >> 5;
      int nf  = (idx & 31) * 4;
      const float* gp = &Bm[(size_t)(k0 + kr) * N + n0 + nf];
      float4 t = *(const float4*)gp;
      if (k0 + 32 < K) __builtin_prefetch(gp + (size_t)32 * N, 0, 0);
      Bs[nf + 0][kr] = (__bf16)t.x;
      Bs[nf + 1][kr] = (__bf16)t.y;
      Bs[nf + 2][kr] = (__bf16)t.z;
      Bs[nf + 3][kr] = (__bf16)t.w;
    }
    if constexpr (sizeof(AT) == 2) async_wait0();
    __syncthreads();

    // ---- fragments + WMMA ----
    V16 af[4];
    const int doff = (lane < 16) ? 0 : 8;
#pragma unroll
    for (int i = 0; i < 4; ++i) {
      int row = wm * 64 + i * 16 + (lane & 15);
      af[i].h[0] = *(const v8bf*)&As[row][doff];
      af[i].h[1] = *(const v8bf*)&As[row][16 + doff];
    }
    const int koff = (lane < 16) ? 0 : 8;
#pragma unroll
    for (int j = 0; j < 2; ++j) {
      V16 bf_;
      int col = wn * 32 + j * 16 + (lane & 15);
      bf_.h[0] = *(const v8bf*)&Bs[col][koff];
      bf_.h[1] = *(const v8bf*)&Bs[col][koff + 16];
#pragma unroll
      for (int i = 0; i < 4; ++i)
        acc[i][j] = wmma_bf16(af[i].v, bf_.v, acc[i][j]);
    }
  }

  // ---- epilogue ----
  const int rbase = (lane & 16) ? 8 : 0;
#pragma unroll
  for (int i = 0; i < 4; ++i) {
#pragma unroll
    for (int j = 0; j < 2; ++j) {
      int col = n0 + wn * 32 + j * 16 + (lane & 15);
      float bv = HAS_BIAS ? bias[col] : 0.0f;
#pragma unroll
      for (int r = 0; r < 8; ++r) {
        int row = m0 + wm * 64 + i * 16 + rbase + r;
        float v = acc[i][j][r];
        if (HAS_BIAS) v += bv;
        if (ACT_SILU) v = fast_silu(v);
        if (HAS_RES)  v += Res[(size_t)row * N + col];
        C[(size_t)row * N + col] = v;
      }
    }
  }
}

// ---------------------------------------------------------------------------
// Fused SwiGLU GEMM: U[M,F](bf16) = silu(A@W1) * (A@W3); block 128x64, BK 32.
// ---------------------------------------------------------------------------
__global__ void __launch_bounds__(256)
ffn_swiglu_kernel(const float* __restrict__ A, const float* __restrict__ W1,
                  const float* __restrict__ W3, __bf16* __restrict__ U,
                  int M, int K, int N) {
  __shared__ __bf16 As[128][48];
  __shared__ __bf16 B1s[64][48];
  __shared__ __bf16 B3s[64][48];

  const int tid  = threadIdx.x;
  const int lane = tid & 31;
  const int wave = tid >> 5;
  const int wm   = wave & 3;       // rows wm*32
  const int wn   = wave >> 2;      // cols wn*32
  const int m0   = blockIdx.y * 128;
  const int n0   = blockIdx.x * 64;

  v8f acc1[2][2] = {};
  v8f acc3[2][2] = {};

  for (int k0 = 0; k0 < K; k0 += 32) {
    __syncthreads();
#pragma unroll
    for (int i = 0; i < 4; ++i) {        // A tile 128x32
      int idx = tid + 256 * i;
      int row = idx >> 3;
      int kf  = (idx & 7) * 4;
      const float* gp = &A[(size_t)(m0 + row) * K + k0 + kf];
      float4 t = *(const float4*)gp;
      if (k0 + 32 < K) __builtin_prefetch(gp + 32, 0, 0);
      v4bf pk;
      pk[0] = (__bf16)t.x; pk[1] = (__bf16)t.y;
      pk[2] = (__bf16)t.z; pk[3] = (__bf16)t.w;
      *(v4bf*)&As[row][kf] = pk;
    }
#pragma unroll
    for (int i = 0; i < 2; ++i) {        // W1/W3 tiles 32x64 each
      int idx = tid + 256 * i;
      int kr  = idx >> 4;
      int nf  = (idx & 15) * 4;
      float4 t1 = *(const float4*)&W1[(size_t)(k0 + kr) * N + n0 + nf];
      float4 t3 = *(const float4*)&W3[(size_t)(k0 + kr) * N + n0 + nf];
      B1s[nf + 0][kr] = (__bf16)t1.x; B1s[nf + 1][kr] = (__bf16)t1.y;
      B1s[nf + 2][kr] = (__bf16)t1.z; B1s[nf + 3][kr] = (__bf16)t1.w;
      B3s[nf + 0][kr] = (__bf16)t3.x; B3s[nf + 1][kr] = (__bf16)t3.y;
      B3s[nf + 2][kr] = (__bf16)t3.z; B3s[nf + 3][kr] = (__bf16)t3.w;
    }
    __syncthreads();

    V16 af[2];
    const int doff = (lane < 16) ? 0 : 8;
#pragma unroll
    for (int i = 0; i < 2; ++i) {
      int row = wm * 32 + i * 16 + (lane & 15);
      af[i].h[0] = *(const v8bf*)&As[row][doff];
      af[i].h[1] = *(const v8bf*)&As[row][16 + doff];
    }
    const int koff = (lane < 16) ? 0 : 8;
#pragma unroll
    for (int j = 0; j < 2; ++j) {
      V16 b1f, b3f;
      int col = wn * 32 + j * 16 + (lane & 15);
      b1f.h[0] = *(const v8bf*)&B1s[col][koff];
      b1f.h[1] = *(const v8bf*)&B1s[col][koff + 16];
      b3f.h[0] = *(const v8bf*)&B3s[col][koff];
      b3f.h[1] = *(const v8bf*)&B3s[col][koff + 16];
#pragma unroll
      for (int i = 0; i < 2; ++i) {
        acc1[i][j] = wmma_bf16(af[i].v, b1f.v, acc1[i][j]);
        acc3[i][j] = wmma_bf16(af[i].v, b3f.v, acc3[i][j]);
      }
    }
  }

  const int rbase = (lane & 16) ? 8 : 0;
#pragma unroll
  for (int i = 0; i < 2; ++i) {
#pragma unroll
    for (int j = 0; j < 2; ++j) {
      int col = n0 + wn * 32 + j * 16 + (lane & 15);
#pragma unroll
      for (int r = 0; r < 8; ++r) {
        int row = m0 + wm * 32 + i * 16 + rbase + r;
        float v1 = acc1[i][j][r];
        float v3 = acc3[i][j][r];
        U[(size_t)row * N + col] = (__bf16)(fast_silu(v1) * v3);
      }
    }
  }
}

// ---------------------------------------------------------------------------
// Flash attention: per block = 128 queries of one (b,h); 8 waves x 16 queries.
// qb/kb/vb are bf16 views [B*L, Dm] (head offset h*64, row stride Dm).
// K tile staged via async global->LDS; V tile transposed through VGPRs.
// ---------------------------------------------------------------------------
__global__ void __launch_bounds__(256)
attn_kernel(const __bf16* __restrict__ qb, const __bf16* __restrict__ kb,
            const __bf16* __restrict__ vb, float* __restrict__ Of) {
  __shared__ __bf16 Kt[32][80];        // [kj][d]   (160B rows)
  __shared__ __bf16 Vt[64][48];        // [d][kj]   (96B rows)
  __shared__ __bf16 Ps[8][16][32];     // per-wave P scratch (64B rows)

  const int tid  = threadIdx.x;
  const int lane = tid & 31;
  const int wave = tid >> 5;
  const int q0   = blockIdx.x * 128;
  const int h    = blockIdx.y;
  const int b    = blockIdx.z;
  const int qw   = q0 + wave * 16;     // wave's first query

  const size_t base = (size_t)b * Lseq * Dm + h * Dhd;
  const __bf16* Qb = qb + base;
  const __bf16* Kb = kb + base;
  const __bf16* Vb = vb + base;

  // Q fragments (A layout, 16x32 each over d)
  V16 a0, a1;
  {
    int row = qw + (lane & 15);
    const __bf16* qr = Qb + (size_t)row * Dm;
    int ds = (lane < 16) ? 0 : 8;
    a0.h[0] = *(const v8bf*)(qr + ds);
    a0.h[1] = *(const v8bf*)(qr + 16 + ds);
    a1.h[0] = *(const v8bf*)(qr + 32 + ds);
    a1.h[1] = *(const v8bf*)(qr + 48 + ds);
  }

  v8f oacc[4] = {};
  float m_r[8], l_r[8];
#pragma unroll
  for (int r = 0; r < 8; ++r) { m_r[r] = -INFINITY; l_r[r] = 0.0f; }

  const int nblk = (q0 + 128) / 32;
  for (int blk = 0; blk < nblk; ++blk) {
    const int kb0 = blk * 32;
    __syncthreads();
    {   // stage K tile (32x64, async to LDS) and V tile transposed (64x32)
      int kj = tid >> 3;
      int dp = (tid & 7) * 8;
      async_copy16(&Kt[kj][dp], Kb + (size_t)(kb0 + kj) * Dm + dp);
      v8bf vv = *(const v8bf*)(Vb + (size_t)(kb0 + kj) * Dm + dp);
#pragma unroll
      for (int i = 0; i < 8; ++i) Vt[dp + i][kj] = vv[i];
      async_wait0();
    }
    __syncthreads();

    if (kb0 <= qw + 15) {
      v8f s0 = {}, s1 = {};
      const int doffA = (lane < 16) ? 0 : 8;
#pragma unroll
      for (int t = 0; t < 2; ++t) {
        V16 b0, b1;
        const __bf16* kr = &Kt[t * 16 + (lane & 15)][0];
        b0.h[0] = *(const v8bf*)(kr + doffA);
        b0.h[1] = *(const v8bf*)(kr + doffA + 16);
        b1.h[0] = *(const v8bf*)(kr + 32 + doffA);
        b1.h[1] = *(const v8bf*)(kr + 32 + doffA + 16);
        v8f z = {};
        v8f st = wmma_bf16(a0.v, b0.v, z);
        st = wmma_bf16(a1.v, b1.v, st);
        if (t == 0) s0 = st; else s1 = st;
      }
      // scale + causal mask
      const int mbase = qw + ((lane & 16) ? 8 : 0);
      const int c0 = kb0 + (lane & 15);
      const int c1 = kb0 + 16 + (lane & 15);
#pragma unroll
      for (int r = 0; r < 8; ++r) {
        float v0 = s0[r] * 0.125f;
        float v1 = s1[r] * 0.125f;
        s0[r] = (c0 <= mbase + r) ? v0 : -INFINITY;
        s1[r] = (c1 <= mbase + r) ? v1 : -INFINITY;
      }
      // online softmax
      float sc[8];
#pragma unroll
      for (int r = 0; r < 8; ++r) {
        float rm = wred_max16(fmaxf(s0[r], s1[r]));
        float mn = fmaxf(m_r[r], rm);
        sc[r] = __expf(m_r[r] - mn);
        m_r[r] = mn;
        float p0 = __expf(s0[r] - mn);
        float p1 = __expf(s1[r] - mn);
        s0[r] = p0; s1[r] = p1;
        l_r[r] = l_r[r] * sc[r] + wred_sum16(p0 + p1);
      }
#pragma unroll
      for (int j = 0; j < 4; ++j)
#pragma unroll
        for (int r = 0; r < 8; ++r) oacc[j][r] *= sc[r];

      // P -> LDS (C layout -> A layout round trip)
      const int mr = (lane & 16) ? 8 : 0;
#pragma unroll
      for (int r = 0; r < 8; ++r) {
        Ps[wave][mr + r][lane & 15]        = (__bf16)s0[r];
        Ps[wave][mr + r][16 + (lane & 15)] = (__bf16)s1[r];
      }
      V16 pa;
      {
        const __bf16* pr = &Ps[wave][lane & 15][0];
        int off = (lane < 16) ? 0 : 8;
        pa.h[0] = *(const v8bf*)(pr + off);
        pa.h[1] = *(const v8bf*)(pr + 16 + off);
      }
      // O += P * V
      const int ko = (lane < 16) ? 0 : 8;
#pragma unroll
      for (int j = 0; j < 4; ++j) {
        V16 vf_;
        const __bf16* vr = &Vt[j * 16 + (lane & 15)][0];
        vf_.h[0] = *(const v8bf*)(vr + ko);
        vf_.h[1] = *(const v8bf*)(vr + ko + 16);
        oacc[j] = wmma_bf16(pa.v, vf_.v, oacc[j]);
      }
    }
  }

  // write O (normalize by 1/l via v_rcp_f32)
#pragma unroll
  for (int j = 0; j < 4; ++j) {
    int col = h * Dhd + j * 16 + (lane & 15);
#pragma unroll
    for (int r = 0; r < 8; ++r) {
      int rowg = qw + ((lane & 16) ? 8 : 0) + r;
      Of[(size_t)(b * Lseq + rowg) * Dm + col] = oacc[j][r] * fast_rcp(l_r[r]);
    }
  }
}

// ---------------------------------------------------------------------------
// Elementwise / small kernels
// ---------------------------------------------------------------------------
__global__ void __launch_bounds__(256)
rmsnorm_kernel(const float* __restrict__ x, const float* __restrict__ w,
               float* __restrict__ out) {
  const int row = blockIdx.x;
  const int tid = threadIdx.x;
  const float* xr = x + (size_t)row * Dm;
  float v[4], s = 0.0f;
#pragma unroll
  for (int i = 0; i < 4; ++i) { v[i] = xr[tid + 256 * i]; s += v[i] * v[i]; }
  s = wred_sum32(s);
  __shared__ float red[8];
  if ((tid & 31) == 0) red[tid >> 5] = s;
  __syncthreads();
  float tot = 0.0f;
#pragma unroll
  for (int i = 0; i < 8; ++i) tot += red[i];
  float sc = rsqrtf(tot * (1.0f / Dm) + 1e-5f);
#pragma unroll
  for (int i = 0; i < 4; ++i)
    out[(size_t)row * Dm + tid + 256 * i] = v[i] * sc * w[tid + 256 * i];
}

__global__ void __launch_bounds__(256)
rope_bf16_kernel(const float* __restrict__ t, const float* __restrict__ cs,
                 const float* __restrict__ sn, __bf16* __restrict__ out) {
  int e = blockIdx.x * 256 + threadIdx.x;
  int col = e & (Dm - 1);
  int dh  = col & (Dhd - 1);
  int l   = (e >> 10) & (Lseq - 1);
  float v = t[e];
  float r = (dh < 32) ? -t[e + 32] : t[e - 32];
  out[e] = (__bf16)(v * cs[l * Dhd + dh] + r * sn[l * Dhd + dh]);
}

__global__ void __launch_bounds__(256)
cvt_bf16_kernel(const float* __restrict__ t, __bf16* __restrict__ out) {
  int e = blockIdx.x * 256 + threadIdx.x;
  out[e] = (__bf16)t[e];
}

__global__ void __launch_bounds__(256)
logits_kernel(const float* __restrict__ hg, const float* __restrict__ w2,
              const float* __restrict__ b2, float* __restrict__ logits) {
  const int lane = threadIdx.x & 31;
  const int row  = blockIdx.x * 8 + (threadIdx.x >> 5);
  const float* hr = hg + (size_t)row * Dm;
  float s = 0.0f;
  for (int j = lane; j < Dm; j += 32) s += hr[j] * w2[j];
  s = wred_sum32(s);
  if (lane == 0) logits[row] = s + b2[0];
}

__global__ void scan_kernel(const float* __restrict__ logits,
                            int* __restrict__ bidx,
                            float* __restrict__ bound_out,
                            float* __restrict__ counts_out) {
  int b = threadIdx.x;
  if (b >= Bsz) return;
  const float* lg = logits + b * Lseq;
  int s = -1, cnt = 0;
  for (int p = 0; p < Lseq; ++p) {
    bool mb = lg[p] > 0.0f;            // sigmoid(x)>0.5 <=> x>0
    int a = (s > 0) ? s : 0;
    bool forced = (!mb) && (p > a) && (((p - a) & (MAXC - 1)) == 0);
    if (mb) s = p;
    if (mb || forced) {
      bidx[b * Lseq + cnt] = p;
      bound_out[b * Lseq + cnt] = (float)p;
      ++cnt;
    }
  }
  counts_out[b] = (float)cnt;
  for (int j = cnt; j < Lseq; ++j) {
    bidx[b * Lseq + j] = Lseq;
    bound_out[b * Lseq + j] = (float)Lseq;
  }
}

__global__ void __launch_bounds__(256)
gather_kernel(const int* __restrict__ bidx, const float* __restrict__ xout,
              float* __restrict__ comp) {
  const int j = blockIdx.x;               // [0, B*L)
  const int b = j >> 11;
  const int tid = threadIdx.x;
  const int pos = bidx[j];
  float* dst = comp + (size_t)j * Dm;
  if (pos < Lseq) {
    const float* src = xout + (size_t)(b * Lseq + pos) * Dm;
#pragma unroll
    for (int i = 0; i < 4; ++i) dst[tid + 256 * i] = src[tid + 256 * i];
  } else {
#pragma unroll
    for (int i = 0; i < 4; ++i) dst[tid + 256 * i] = 0.0f;
  }
}

// ---------------------------------------------------------------------------
// Host side
// ---------------------------------------------------------------------------
extern "C" void kernel_launch(void* const* d_in, const int* in_sizes, int n_in,
                              void* d_out, int out_size, void* d_ws, size_t ws_size,
                              hipStream_t stream) {
  (void)in_sizes; (void)n_in; (void)out_size; (void)ws_size;

  // Inputs in setup_inputs() insertion order (params flattened in order).
  const float* x_in = (const float*)d_in[0];
  const float* cosp = (const float*)d_in[1];
  const float* sinp = (const float*)d_in[2];
  auto LW = [&](int layer, int which) -> const float* {
    // which: 0 wq, 1 wk, 2 wv, 3 wo, 4 w1, 5 w3, 6 w2, 7 attn_norm, 8 mlp_norm
    return (const float*)d_in[3 + layer * 9 + which];
  };
  const float* norm_w = (const float*)d_in[21];
  const float* g_w1 = (const float*)d_in[22];
  const float* g_b1 = (const float*)d_in[23];
  const float* g_w2 = (const float*)d_in[24];
  const float* g_b2 = (const float*)d_in[25];

  // Workspace layout (bytes)
  char* ws = (char*)d_ws;
  const size_t SZ_F32 = (size_t)Mrows * Dm * 4;   // 16 MB
  const size_t SZ_BF  = (size_t)Mrows * Dm * 2;   // 8 MB
  float*  xcur  = (float*)(ws + 0);
  float*  h     = (float*)(ws + SZ_F32);
  float*  qf    = (float*)(ws + 2 * SZ_F32);
  float*  kf    = (float*)(ws + 3 * SZ_F32);
  __bf16* qb    = (__bf16*)(ws + 4 * SZ_F32);
  __bf16* kbuf  = (__bf16*)(ws + 4 * SZ_F32 + SZ_BF);
  __bf16* vbuf  = (__bf16*)(ws + 4 * SZ_F32 + 2 * SZ_BF);
  float*  vf    = (float*)(ws + 4 * SZ_F32 + 3 * SZ_BF);
  float*  of    = vf;                                 // alias (vf dead)
  __bf16* u     = (__bf16*)qf;                        // alias: u (33.5MB) over qf+kf
  float*  logits = (float*)(ws + 5 * SZ_F32 + 3 * SZ_BF);
  int*    bidx   = (int*)(ws + 5 * SZ_F32 + 3 * SZ_BF + Mrows * 4);

  float* out   = (float*)d_out;
  float* xout  = out;                                 // [B,L,D]
  float* comp  = out + (size_t)Mrows * Dm;            // [B,L,D]
  float* bound = out + 2 * (size_t)Mrows * Dm;        // [B,L]
  float* cnts  = bound + Mrows;                       // [B]

  const dim3 blk(256);
  const dim3 gGemmD(Dm / 128, Mrows / 128);           // (8, 32)
  const dim3 gFfn(Ff / 64, Mrows / 128);              // (64, 32)
  const dim3 gAttn(Lseq / 128, Hh, Bsz);              // (16, 16, 2)
  const int  gElem = (Mrows * Dm) / 256;              // 16384

  hipMemcpyAsync(xcur, x_in, SZ_F32, hipMemcpyDeviceToDevice, stream);

  for (int layer = 0; layer < 2; ++layer) {
    const float* wq = LW(layer, 0); const float* wk = LW(layer, 1);
    const float* wv = LW(layer, 2); const float* wo = LW(layer, 3);
    const float* w1 = LW(layer, 4); const float* w3 = LW(layer, 5);
    const float* w2 = LW(layer, 6);
    const float* an = LW(layer, 7); const float* mn = LW(layer, 8);

    rmsnorm_kernel<<<Mrows, blk, 0, stream>>>(xcur, an, h);
    gemm_kernel<float, false, false, false><<<gGemmD, blk, 0, stream>>>(
        h, wq, nullptr, nullptr, qf, Mrows, Dm, Dm);
    gemm_kernel<float, false, false, false><<<gGemmD, blk, 0, stream>>>(
        h, wk, nullptr, nullptr, kf, Mrows, Dm, Dm);
    gemm_kernel<float, false, false, false><<<gGemmD, blk, 0, stream>>>(
        h, wv, nullptr, nullptr, vf, Mrows, Dm, Dm);
    rope_bf16_kernel<<<gElem, blk, 0, stream>>>(qf, cosp, sinp, qb);
    rope_bf16_kernel<<<gElem, blk, 0, stream>>>(kf, cosp, sinp, kbuf);
    cvt_bf16_kernel<<<gElem, blk, 0, stream>>>(vf, vbuf);
    attn_kernel<<<gAttn, blk, 0, stream>>>(qb, kbuf, vbuf, of);
    gemm_kernel<float, false, false, true><<<gGemmD, blk, 0, stream>>>(
        of, wo, nullptr, xcur, xcur, Mrows, Dm, Dm);
    rmsnorm_kernel<<<Mrows, blk, 0, stream>>>(xcur, mn, h);
    ffn_swiglu_kernel<<<gFfn, blk, 0, stream>>>(h, w1, w3, u, Mrows, Dm, Ff);
    gemm_kernel<__bf16, false, false, true><<<gGemmD, blk, 0, stream>>>(
        u, w2, nullptr, xcur, xcur, Mrows, Dm, Ff);
  }

  rmsnorm_kernel<<<Mrows, blk, 0, stream>>>(xcur, norm_w, xout);
  gemm_kernel<float, true, true, false><<<gGemmD, blk, 0, stream>>>(
      xout, g_w1, g_b1, nullptr, h, Mrows, Dm, Dm);
  logits_kernel<<<Mrows / 8, blk, 0, stream>>>(h, g_w2, g_b2, logits);
  scan_kernel<<<1, 32, 0, stream>>>(logits, bidx, bound, cnts);
  gather_kernel<<<Mrows, blk, 0, stream>>>(bidx, xout, comp);
}
